// DenseEnergyLoss_83099027243295
// MI455X (gfx1250) — matrix operand order repaired and checked by reading, not compiled, not run on hardware
//
#include <hip/hip_runtime.h>
#include <hip/hip_bf16.h>
#include <cstdint>

typedef _Float16 half_t;
typedef __attribute__((ext_vector_type(16))) _Float16 v16h;
typedef __attribute__((ext_vector_type(8)))  float    v8f;
typedef __attribute__((ext_vector_type(2)))  float    v2f;

#define NN   4
#define KK   21
#define PP   4096          // 64*64 downsampled pixels
#define KPAD 32            // K padded for WMMA f16 (16x16x32)
#define QCHUNK 256
#define QSLICES 8
#define PBLOCKS 32         // P / 128
#define L2E  1.4426950408889634f
#define SQRT_L2E 1.2011224087864498f   // sqrt(log2(e)): both operands scaled

union AFragU { v16h h; uint32_t u[8]; };

// ---------------------------------------------------------------------------
// Kernel 1: downsample + feature/gate preparation.
// Augmented features so that  fa . fb = log2(e) * (-1/2)*||f_p - f_q||^2 :
//   featA[n][p][8] = {c*f0..c*f4, -0.5*L2E*|f|^2, 1, 0}   (c = sqrt(log2 e))
//   featB[n][p][8] = {c*f0..c*f4, 1, -0.5*L2E*|f|^2, 0}
// segH[n][p][32]     = fp16 seg_m (k>=21 zero)
// segGateH[n][p][32] = fp16 gate*seg_m
// ---------------------------------------------------------------------------
__global__ __launch_bounds__(256) void prep_kernel(
    const float* __restrict__ images, const float* __restrict__ segs,
    const float* __restrict__ rois,   const float* __restrict__ label,
    float* __restrict__ featA, float* __restrict__ featB,
    half_t* __restrict__ segH, half_t* __restrict__ segGateH) {
  int id = blockIdx.x * 256 + threadIdx.x;       // 0 .. N*P-1
  int n = id >> 12;
  int p = id & (PP - 1);
  int y = p >> 6, x = p & 63;
  int y2 = y << 1, x2 = x << 1;

  float f[5];
  f[0] = (float)x * (1.0f / 40.0f);              // SIGMA_XY*SCALE = 40
  f[1] = (float)y * (1.0f / 40.0f);
#pragma unroll
  for (int c = 0; c < 3; ++c)
    f[2 + c] = images[(((size_t)n * 3 + c) * 128 + y2) * 128 + x2] * (1.0f / 15.0f);
  float sq = 0.0f;
#pragma unroll
  for (int c = 0; c < 5; ++c) sq += f[c] * f[c];
  float nsq = -0.5f * L2E * sq;

  size_t fb = (size_t)id * 8;
#pragma unroll
  for (int c = 0; c < 5; ++c) {
    float v = f[c] * SQRT_L2E;
    featA[fb + c] = v;
    featB[fb + c] = v;
  }
  featA[fb + 5] = nsq;  featA[fb + 6] = 1.0f;  featA[fb + 7] = 0.0f;
  featB[fb + 5] = 1.0f; featB[fb + 6] = nsq;   featB[fb + 7] = 0.0f;

  float roi = rois[((size_t)n * 128 + y2) * 128 + x2];
  float lab = label[((size_t)n * 128 + y2) * 128 + x2];
  bool unl = ((int)lab == 255);

  float s[KK];
  float mx = -1e30f;
#pragma unroll
  for (int k = 0; k < KK; ++k) {
    size_t b = (((size_t)n * KK + k) * 128 + y2) * 128 + x2;
    float v = 0.25f * (segs[b] + segs[b + 1] + segs[b + 128] + segs[b + 129]);
    s[k] = v;
    mx = fmaxf(mx, v);
  }
  float gate = unl ? 1.0f : (roi - mx);
  gate = fmaxf(gate, 0.0f);

  size_t hb = (size_t)id * KPAD;
#pragma unroll
  for (int k = 0; k < KPAD; ++k) {
    float sm = (k < KK) ? s[k] * roi : 0.0f;
    segH[hb + k]     = (half_t)sm;
    segGateH[hb + k] = (half_t)(gate * sm);
  }
}

// ---------------------------------------------------------------------------
// Kernel 2: fully-fused tile loop. Per 16x16 (p,q) tile:
//   2x V_WMMA_F32_16X16X4_F32  -> exp2 argument (clamped Gaussian exponent)
//   1x V_WMMA_F32_16X16X32_F16 -> K=21 seg dot (gate folded into A operand)
//   8x (v_min, v_exp, v_fma)   -> scalar accumulation
// grid = (QSLICES, PBLOCKS, N), block = 256 (8 waves, one 16-row p-tile each)
// ---------------------------------------------------------------------------
__global__ __launch_bounds__(256) void main_kernel(
    const float* __restrict__ featA, const float* __restrict__ featB,
    const half_t* __restrict__ segH, const half_t* __restrict__ segGateH,
    float* __restrict__ partials) {
  __shared__ uint32_t segQ[QCHUNK * 17];   // padded stride: conflict-free B reads
  __shared__ float    featQ[QCHUNK * 9];   // padded stride
  __shared__ float    wsum[8];

  const int n    = blockIdx.z;
  const int pb   = blockIdx.y;
  const int qs   = blockIdx.x;
  const int wave = threadIdx.x >> 5;
  const int lane = threadIdx.x & 31;
  const int hi   = (lane >= 16) ? 1 : 0;
  const int l16  = lane & 15;

  const int pTileBase = pb * 128 + wave * 16;

  // ---- f16 A fragment (gate*seg), documented 16-bit A layout:
  // lane holds row M = lane%16; VGPR j holds K pair k0 = 2j + 8*(j>=4) + 8*(lane>=16)
  AFragU aF;
  {
    const size_t rowBase = ((size_t)n * PP + pTileBase + l16) * KPAD;
#pragma unroll
    for (int j = 0; j < 8; ++j) {
      int k0 = 2 * j + ((j >= 4) ? 8 : 0) + (hi ? 8 : 0);
      aF.u[j] = *(const uint32_t*)(segGateH + rowBase + k0);
    }
  }

  // ---- f32 A fragments (augmented p-features), 16x4 layout:
  // lane half selects K-half, VGPR index selects K within half.
  v2f aG0, aG1;
  {
    const float* fa = featA + ((size_t)n * PP + pTileBase + l16) * 8;
    aG0.x = fa[2 * hi];     aG0.y = fa[2 * hi + 1];
    aG1.x = fa[4 + 2 * hi]; aG1.y = fa[4 + 2 * hi + 1];
  }

  float partial = 0.0f;

  for (int chunk = 0; chunk < 2; ++chunk) {
    const int qBase = qs * (2 * QCHUNK) + chunk * QCHUNK;
    __syncthreads();
    // stage f16 seg (16 dwords/row) -- coalesced uint4 loads, padded LDS rows
    {
      const uint4* src = (const uint4*)(segH + ((size_t)n * PP + qBase) * KPAD);
#pragma unroll
      for (int pass = 0; pass < 4; ++pass) {
        int e = pass * 256 + threadIdx.x;          // 0..1023
        int q = e >> 2, i4 = e & 3;
        uint4 v = src[q * 4 + i4];
        int d = q * 17 + i4 * 4;
        segQ[d + 0] = v.x; segQ[d + 1] = v.y;
        segQ[d + 2] = v.z; segQ[d + 3] = v.w;
      }
      const float* fsrc = featB + ((size_t)n * PP + qBase) * 8;
#pragma unroll
      for (int e = threadIdx.x; e < QCHUNK * 8; e += 256)
        featQ[(e >> 3) * 9 + (e & 7)] = fsrc[e];
    }
    // prefetch next chunk while computing this one
    __builtin_prefetch(segH + ((size_t)n * PP + qBase + QCHUNK) * KPAD, 0, 0);
    __syncthreads();

    for (int t = 0; t < 16; ++t) {
      const int qrow = t * 16 + l16;

      // f16 B fragment (seg): lane holds col N = lane%16;
      // VGPR j holds K pair 2j + 8*(j>=4) + 8*(lane>=16)
      AFragU bF;
#pragma unroll
      for (int j = 0; j < 8; ++j)
        bF.u[j] = segQ[qrow * 17 + j + ((j >= 4) ? 4 : 0) + 4 * hi];

      // f32 B fragments (augmented q-features), 4x16 layout mirrors A
      v2f bG0, bG1;
      bG0.x = featQ[qrow * 9 + 2 * hi];     bG0.y = featQ[qrow * 9 + 2 * hi + 1];
      bG1.x = featQ[qrow * 9 + 4 + 2 * hi]; bG1.y = featQ[qrow * 9 + 4 + 2 * hi + 1];

      // exponent tile: e[j] = log2(e) * (-1/2) d2(p_Mj, q_N)
      v8f ex = {};
      ex = __builtin_amdgcn_wmma_f32_16x16x4_f32(
          false, aG0, false, bG0, (short)0, ex, false, false);
      ex = __builtin_amdgcn_wmma_f32_16x16x4_f32(
          false, aG1, false, bG1, (short)0, ex, false, false);

      // seg-dot tile: gs[j] = sum_k gate*seg[p,k] * seg[q,k]
      v8f gs = {};
      gs = __builtin_amdgcn_wmma_f32_16x16x32_f16(
          false, aF.h, false, bF.h, (short)0, gs, false, false);

#pragma unroll
      for (int j = 0; j < 8; ++j) {
        float arg = fminf(ex[j], 0.0f);          // d2 clamped at 0
        partial = fmaf(__builtin_amdgcn_exp2f(arg), gs[j], partial);
      }
    }
  }

  // ---- deterministic block reduction
#pragma unroll
  for (int off = 16; off > 0; off >>= 1)
    partial += __shfl_down(partial, off, 32);
  if (lane == 0) wsum[wave] = partial;
  __syncthreads();
  if (threadIdx.x == 0) {
    float s = 0.0f;
#pragma unroll
    for (int w = 0; w < 8; ++w) s += wsum[w];
    partials[((size_t)n * PBLOCKS + pb) * QSLICES + qs] = s;
  }
}

// ---------------------------------------------------------------------------
// Kernel 3: final reduction + scaling
// ---------------------------------------------------------------------------
__global__ __launch_bounds__(256) void reduce_kernel(
    const float* __restrict__ partials, float* __restrict__ out) {
  __shared__ float s[256];
  int t = threadIdx.x;
  float v = partials[t] + partials[t + 256] + partials[t + 512] + partials[t + 768];
  s[t] = v;
  __syncthreads();
#pragma unroll
  for (int off = 128; off > 0; off >>= 1) {
    if (t < off) s[t] += s[t + off];
    __syncthreads();
  }
  if (t == 0)
    out[0] = s[0] * (-0.5f * 0.1f / (float)(NN * PP));  // -0.5 * WEIGHT / (N*P)
}

// ---------------------------------------------------------------------------
extern "C" void kernel_launch(void* const* d_in, const int* in_sizes, int n_in,
                              void* d_out, int out_size, void* d_ws, size_t ws_size,
                              hipStream_t stream) {
  const float* images = (const float*)d_in[0];   // N,3,128,128
  const float* segs   = (const float*)d_in[1];   // N,21,128,128
  const float* rois   = (const float*)d_in[2];   // N,128,128
  const float* label  = (const float*)d_in[3];   // N,1,128,128

  // workspace carve-out
  float*  featA    = (float*)d_ws;                                // N*P*8 f32
  float*  featB    = featA + (size_t)NN * PP * 8;                 // N*P*8 f32
  half_t* segH     = (half_t*)(featB + (size_t)NN * PP * 8);      // N*P*32 f16
  half_t* segGateH = segH + (size_t)NN * PP * KPAD;               // N*P*32 f16
  float*  partials = (float*)(segGateH + (size_t)NN * PP * KPAD); // 1024 f32

  prep_kernel<<<NN * PP / 256, 256, 0, stream>>>(images, segs, rois, label,
                                                 featA, featB, segH, segGateH);
  main_kernel<<<dim3(QSLICES, PBLOCKS, NN), 256, 0, stream>>>(
      featA, featB, segH, segGateH, partials);
  reduce_kernel<<<1, 256, 0, stream>>>(partials, (float*)d_out);
}
